// CSSMUpdateBlock_69853348102331
// MI455X (gfx1250) — compile-verified, matrix operand order checked by reading, not run on hardware
//
#include <hip/hip_runtime.h>
#include <hip/hip_bf16.h>

// ---------------------------------------------------------------------------
// Problem constants (from the reference)
// ---------------------------------------------------------------------------
#define BB   4
#define TT   32
#define NN   2048
#define HH   256
#define DIN_RAW 517          // D + DC + DF + 1
#define DIN_PAD 544          // padded to multiple of 32 for WMMA K tiles
#define NTOT 2112            // N + NV
#define MROWS (BB*TT*NN)     // 262144
#define MTOT  (BB*TT*NTOT)   // 270336

typedef unsigned short u16;
typedef __attribute__((ext_vector_type(16))) __bf16 v16bf;
typedef __attribute__((ext_vector_type(8)))  float  v8f;
typedef __attribute__((ext_vector_type(4)))  int    i4;

#if __has_builtin(__builtin_amdgcn_global_load_async_to_lds_b128)
#define HAVE_ASYNC_LDS 1
#else
#define HAVE_ASYNC_LDS 0
#endif

// ---------------------------------------------------------------------------
// Helpers
// ---------------------------------------------------------------------------
__device__ __forceinline__ u16 f32_to_bf16(float f) {
    unsigned int u = __builtin_bit_cast(unsigned int, f);
    unsigned int lsb = (u >> 16) & 1u;
    u += 0x7fffu + lsb;                 // round to nearest even
    return (u16)(u >> 16);
}
__device__ __forceinline__ float bf16_to_f32(u16 x) {
    unsigned int u = ((unsigned int)x) << 16;
    return __builtin_bit_cast(float, u);
}
__device__ __forceinline__ float gelu_f(float x) {
    const float c = 0.7978845608028654f;
    float x3 = x * x * x;
    float t = tanhf(c * (x + 0.044715f * x3));
    return 0.5f * x * (1.0f + t);
}
__device__ __forceinline__ float sigmoid_f(float x) {
    return 1.0f / (1.0f + expf(-x));
}

#if HAVE_ASYNC_LDS
// 16-byte async DMA: global -> LDS, tracked by ASYNCcnt.
// Builtin signature (per hipcc diagnostic): (int4 as(1)*, int4 as(3)*, Ii, Ii).
// Address-space casts via integers: LDS generic addresses carry the LDS byte
// offset in the low 32 bits; global generic == addrspace(1) bit pattern.
__device__ __forceinline__ void async_copy16(const u16* g, const u16* l) {
    __builtin_amdgcn_global_load_async_to_lds_b128(
        (__attribute__((address_space(1))) i4*)(unsigned long long)g,
        (__attribute__((address_space(3))) i4*)(unsigned int)(unsigned long long)l,
        0, 0);
}
__device__ __forceinline__ void wait_async0() {
#if __has_builtin(__builtin_amdgcn_s_wait_asynccnt)
    __builtin_amdgcn_s_wait_asynccnt(0);
#else
    asm volatile("s_wait_asynccnt 0" ::: "memory");
#endif
}
#endif

// Load a 16x32 bf16 fragment from LDS (row stride 40 u16 = 80 B).
// Contiguous k order (same for A and B): lanes 0-15 hold K=0..15 of row (lane),
// lanes 16-31 hold K=16..31 of row (lane-16): 32 contiguous bytes per lane.
__device__ __forceinline__ v16bf ld_frag(const u16* __restrict__ base, int lane) {
    const u16* p = base + (lane & 15) * 40 + (lane >> 4) * 16;
    union { uint4 q[2]; v16bf v; } f;
    f.q[0] = *(const uint4*)(p);
    f.q[1] = *(const uint4*)(p + 8);
    return f.v;
}

// ---------------------------------------------------------------------------
// Generic bf16 WMMA GEMM:  C[M x N] = epilogue( A[M x K] * Bt[N x K]^T )
//   A row-major bf16, Bt row-major bf16 (B transposed: columns contiguous).
//   Block tile 128x128, BK=32, 8 waves (wave32) as 4(M) x 2(N); each wave owns
//   a 32x64 tile = 2x4 v_wmma_f32_16x16x32_bf16 accumulators.
//   Double-buffered LDS staging, async global->LDS DMA when available.
//   Requires M%128==0, N%128==0, K%32==0 (true for all calls here).
// epilogue: v = acc + bias[col]; if(act) v = gelu(v); if(resid) v += resid
// outputs: optional fp32 Cf and/or bf16 Cb.
// ---------------------------------------------------------------------------
__global__ __launch_bounds__(256)
void gemm_bf16_wmma(const u16* __restrict__ A, const u16* __restrict__ Bt,
                    const float* __restrict__ bias, const float* __restrict__ resid,
                    float* __restrict__ Cf, u16* __restrict__ Cb,
                    int M, int N, int K, int act)
{
    __shared__ __align__(16) u16 sA[2][128 * 40];
    __shared__ __align__(16) u16 sB[2][128 * 40];

    const int tid  = threadIdx.x;
    const int lane = tid & 31;
    const int wave = tid >> 5;
    const int wm   = wave >> 1;     // 0..3 -> 32 rows each
    const int wn   = wave & 1;      // 0..1 -> 64 cols each
    const int rowBase = blockIdx.y * 128;
    const int colBase = blockIdx.x * 128;
    const int KT = K >> 5;

    v8f acc[2][4] = {};

    // Per-thread chunk map: tile is 128 rows x 32 bf16 = 512 x 16B chunks for
    // A and likewise for B; each of 256 threads moves 2 A-chunks + 2 B-chunks.
#if HAVE_ASYNC_LDS
    auto issueAsync = [&](int buf, int kt) {
        const int k0 = kt << 5;
#pragma unroll
        for (int p = 0; p < 2; ++p) {
            int idx = tid + p * 256;
            int r = idx >> 2, c = idx & 3;
            async_copy16(A  + (size_t)(rowBase + r) * K + k0 + c * 8,
                         &sA[buf][r * 40 + c * 8]);
            async_copy16(Bt + (size_t)(colBase + r) * K + k0 + c * 8,
                         &sB[buf][r * 40 + c * 8]);
        }
    };
#else
    uint4 ra[2], rb[2];
    auto loadG = [&](int kt) {
        const int k0 = kt << 5;
#pragma unroll
        for (int p = 0; p < 2; ++p) {
            int idx = tid + p * 256;
            int r = idx >> 2, c = idx & 3;
            ra[p] = *(const uint4*)(A  + (size_t)(rowBase + r) * K + k0 + c * 8);
            rb[p] = *(const uint4*)(Bt + (size_t)(colBase + r) * K + k0 + c * 8);
        }
    };
    auto storeL = [&](int buf) {
#pragma unroll
        for (int p = 0; p < 2; ++p) {
            int idx = tid + p * 256;
            int r = idx >> 2, c = idx & 3;
            *(uint4*)(&sA[buf][r * 40 + c * 8]) = ra[p];
            *(uint4*)(&sB[buf][r * 40 + c * 8]) = rb[p];
        }
    };
#endif

    auto compute = [&](int buf) {
        v16bf a[2], b[4];
#pragma unroll
        for (int i = 0; i < 2; ++i)
            a[i] = ld_frag(&sA[buf][(wm * 32 + i * 16) * 40], lane);
#pragma unroll
        for (int j = 0; j < 4; ++j)
            b[j] = ld_frag(&sB[buf][(wn * 64 + j * 16) * 40], lane);
#pragma unroll
        for (int i = 0; i < 2; ++i)
#pragma unroll
            for (int j = 0; j < 4; ++j)
                acc[i][j] = __builtin_amdgcn_wmma_f32_16x16x32_bf16(
                    false, a[i], false, b[j], (short)0, acc[i][j], false, false);
    };

#if HAVE_ASYNC_LDS
    issueAsync(0, 0);
    wait_async0();
    __syncthreads();
    for (int kt = 0; kt < KT; ++kt) {
        const int buf = kt & 1;
        if (kt + 1 < KT) issueAsync(buf ^ 1, kt + 1);
        if (kt + 2 < KT) {   // speculative prefetch two tiles ahead
            __builtin_prefetch(A  + (size_t)(rowBase + (tid & 127)) * K + ((kt + 2) << 5), 0, 1);
            __builtin_prefetch(Bt + (size_t)(colBase + (tid & 127)) * K + ((kt + 2) << 5), 0, 1);
        }
        compute(buf);
        if (kt + 1 < KT) {
            wait_async0();       // my DMA into buf^1 landed
            __syncthreads();     // everyone's landed; reads of buf done
        }
    }
#else
    loadG(0);
    storeL(0);
    for (int kt = 0; kt < KT; ++kt) {
        __syncthreads();
        const int buf = kt & 1;
        if (kt + 1 < KT) loadG(kt + 1);
        if (kt + 2 < KT) {
            __builtin_prefetch(A  + (size_t)(rowBase + (tid & 127)) * K + ((kt + 2) << 5), 0, 1);
            __builtin_prefetch(Bt + (size_t)(colBase + (tid & 127)) * K + ((kt + 2) << 5), 0, 1);
        }
        compute(buf);
        if (kt + 1 < KT) {
            __syncthreads();
            storeL((kt + 1) & 1);
        }
    }
#endif

    // ---- epilogue (C/D layout: VGPR r -> M = r + 8*(lane>=16), N = lane&15)
    const int cRow = lane >> 4;
    const int cCol = lane & 15;
#pragma unroll
    for (int i = 0; i < 2; ++i)
#pragma unroll
        for (int j = 0; j < 4; ++j) {
#pragma unroll
            for (int r = 0; r < 8; ++r) {
                int row = rowBase + wm * 32 + i * 16 + cRow * 8 + r;
                int col = colBase + wn * 64 + j * 16 + cCol;
                float v = acc[i][j][r];
                if (bias) v += bias[col];
                if (act == 1) v = gelu_f(v);
                size_t o = (size_t)row * N + col;
                if (resid) v += resid[o];
                if (Cf) Cf[o] = v;
                if (Cb) Cb[o] = f32_to_bf16(v);
            }
        }
}

// ---------------------------------------------------------------------------
// Pack concat([track(256), corr(196), flow(64), vis(1)]) -> bf16, K padded 544
// ---------------------------------------------------------------------------
__global__ void pack_xcat(const float* __restrict__ tr, const float* __restrict__ co,
                          const float* __restrict__ fl, const float* __restrict__ vi,
                          u16* __restrict__ xcat)
{
    size_t idx = (size_t)blockIdx.x * 256 + threadIdx.x;
    const size_t total = (size_t)MROWS * DIN_PAD;
    if (idx >= total) return;
    int row = (int)(idx / DIN_PAD);
    int c   = (int)(idx % DIN_PAD);
    float v;
    if      (c < 256) v = tr[(size_t)row * 256 + c];
    else if (c < 452) v = co[(size_t)row * 196 + (c - 256)];
    else if (c < 516) v = fl[(size_t)row * 64  + (c - 452)];
    else if (c == 516) v = vi[row];
    else v = 0.0f;
    xcat[idx] = f32_to_bf16(v);
}

// W (K x N, fp32, row-major) -> Wt (N x Kpad, bf16, row-major), zero-padded K
__global__ void wtrans(const float* __restrict__ W, u16* __restrict__ Wt,
                       int K, int N, int Kpad)
{
    size_t idx = (size_t)blockIdx.x * 256 + threadIdx.x;
    if (idx >= (size_t)N * Kpad) return;
    int nn = (int)(idx / Kpad);
    int k  = (int)(idx % Kpad);
    Wt[idx] = (k < K) ? f32_to_bf16(W[(size_t)k * N + nn]) : (u16)0;
}

// xv[b,t,n,h] = (n<2048) ? x[b,t,n,h] : virtual_tokens[n-2048,h]
__global__ void assemble_xv(const float* __restrict__ xbuf, const float* __restrict__ vt,
                            float* __restrict__ xv)
{
    size_t idx = (size_t)blockIdx.x * 256 + threadIdx.x;
    const size_t total = (size_t)MTOT * HH;
    if (idx >= total) return;
    size_t bt = idx / ((size_t)NTOT * HH);
    int rem = (int)(idx % ((size_t)NTOT * HH));
    int n = rem / HH, h = rem % HH;
    xv[idx] = (n < NN) ? xbuf[(bt * NN + n) * HH + h]
                       : vt[(size_t)(n - NN) * HH + h];
}

// Gated opponent E/I scan over T (sequential, in place: xv += (e - i)).
__global__ void cssm_scan(float* __restrict__ xv,
                          const float* __restrict__ decay_e, const float* __restrict__ decay_i,
                          const float* __restrict__ w_e, const float* __restrict__ w_i)
{
    size_t idx = (size_t)blockIdx.x * 256 + threadIdx.x;
    const size_t total = (size_t)BB * NTOT * HH;
    if (idx >= total) return;
    int h = (int)(idx % HH);
    int n = (int)((idx / HH) % NTOT);
    int b = (int)(idx / ((size_t)NTOT * HH));

    const float ae = sigmoid_f(decay_e[h]);
    const float ai = sigmoid_f(decay_i[h]);
    const float we = w_e[h];
    const float wi = w_i[h];

    size_t base   = ((size_t)b * TT * NTOT + n) * HH + h;
    const size_t stride = (size_t)NTOT * HH;
    float e = 0.0f, ii = 0.0f;
#pragma unroll 4
    for (int t = 0; t < TT; ++t) {
        float xt = xv[base + (size_t)t * stride];
        e  = ae * e  + (1.0f - ae) * (we * xt);
        ii = ai * ii + (1.0f - ai) * (wi * e);
        xv[base + (size_t)t * stride] = xt + (e - ii);
    }
}

// LayerNorm over H=256, one block (256 threads) per row; optional residual;
// rows with n >= Nkeep dropped from the compacted output (strips virt tokens).
__global__ __launch_bounds__(256)
void ln_kernel(const float* __restrict__ in, const float* __restrict__ resid,
               const float* __restrict__ gamma, const float* __restrict__ beta,
               float* __restrict__ outF, u16* __restrict__ outB,
               int Nin, int Nkeep)
{
    const int row = blockIdx.x;
    const int h = threadIdx.x;
    const int n = row % Nin;

    float x = in[(size_t)row * HH + h];
    if (resid) x += resid[(size_t)row * HH + h];

    __shared__ float red[256];
    red[h] = x;
    __syncthreads();
#pragma unroll
    for (int s = 128; s > 0; s >>= 1) {
        if (h < s) red[h] += red[h + s];
        __syncthreads();
    }
    float mean = red[0] * (1.0f / HH);
    __syncthreads();
    float d = x - mean;
    red[h] = d * d;
    __syncthreads();
#pragma unroll
    for (int s = 128; s > 0; s >>= 1) {
        if (h < s) red[h] += red[h + s];
        __syncthreads();
    }
    float var = red[0] * (1.0f / HH);
    float y = d * rsqrtf(var + 1e-6f) * gamma[h] + beta[h];

    if (n < Nkeep) {
        size_t orow = (size_t)(row / Nin) * Nkeep + n;
        if (outF) outF[orow * HH + h] = y;
        if (outB) outB[orow * HH + h] = f32_to_bf16(y);
    }
}

// Tiny heads: delta (2 cols) + vis (1 col) — too narrow for WMMA.
__global__ void heads_kernel(const u16* __restrict__ xc,
                             const float* __restrict__ Wd, const float* __restrict__ bd,
                             const float* __restrict__ Wv, const float* __restrict__ bv,
                             float* __restrict__ delta, float* __restrict__ vis)
{
    int row = blockIdx.x * blockDim.x + threadIdx.x;
    if (row >= MROWS) return;
    const u16* x = xc + (size_t)row * HH;
    float d0 = 0.f, d1 = 0.f, v = 0.f;
#pragma unroll 8
    for (int h = 0; h < HH; ++h) {
        float xv = bf16_to_f32(x[h]);
        d0 += xv * Wd[h * 2 + 0];
        d1 += xv * Wd[h * 2 + 1];
        v  += xv * Wv[h];
    }
    delta[(size_t)row * 2 + 0] = d0 + bd[0];
    delta[(size_t)row * 2 + 1] = d1 + bd[1];
    vis[row] = v + bv[0];
}

// ---------------------------------------------------------------------------
// Launch
// ---------------------------------------------------------------------------
extern "C" void kernel_launch(void* const* d_in, const int* in_sizes, int n_in,
                              void* d_out, int out_size, void* d_ws, size_t ws_size,
                              hipStream_t stream)
{
    (void)in_sizes; (void)n_in; (void)out_size; (void)ws_size;

    const float* track = (const float*)d_in[0];
    const float* corr  = (const float*)d_in[1];
    const float* flow  = (const float*)d_in[2];
    const float* vis   = (const float*)d_in[3];
    const float* W_in  = (const float*)d_in[4];
    const float* b_in  = (const float*)d_in[5];
    const float* vtok  = (const float*)d_in[6];
    const float* dec_e = (const float*)d_in[7];
    const float* dec_i = (const float*)d_in[8];
    const float* w_e   = (const float*)d_in[9];
    const float* w_i   = (const float*)d_in[10];
    const float* ln1_s = (const float*)d_in[11];
    const float* ln1_b = (const float*)d_in[12];
    const float* W_up  = (const float*)d_in[13];
    const float* b_up  = (const float*)d_in[14];
    const float* W_dn  = (const float*)d_in[15];
    const float* b_dn  = (const float*)d_in[16];
    const float* ln2_s = (const float*)d_in[17];
    const float* ln2_b = (const float*)d_in[18];
    const float* W_dl  = (const float*)d_in[19];
    const float* b_dl  = (const float*)d_in[20];
    const float* W_vs  = (const float*)d_in[21];
    const float* b_vs  = (const float*)d_in[22];
    const float* W_tr  = (const float*)d_in[23];
    const float* b_tr  = (const float*)d_in[24];

    float* out       = (float*)d_out;
    float* out_delta = out;                        // (B,T,N,2)
    float* out_vis   = out + (size_t)MROWS * 2;    // (B,T,N,1)
    float* out_track = out_vis + (size_t)MROWS;    // (B,T,N,256)

    // ---- workspace suballocation (lifetime-based aliasing) ----
    char* ws = (char*)d_ws;
    size_t off = 0;
    auto alloc = [&](size_t bytes) -> char* {
        off = (off + 255) & ~(size_t)255;
        char* p = ws + off;
        off += bytes;
        return p;
    };
    // region 0: Xcat (pack->gemm1), later reused as hdn (gemm3->ln2)
    char* r0 = alloc((size_t)MROWS * DIN_PAD * sizeof(u16));
    // region 1: xbuf fp32 (gemm1->assemble), later reused as xc_bf (ln2->end)
    char* r1 = alloc((size_t)MROWS * HH * sizeof(float));
    float* xv     = (float*)alloc((size_t)MTOT * HH * sizeof(float));
    float* xln1   = (float*)alloc((size_t)MTOT * HH * sizeof(float));
    u16*   xln1b  = (u16*)  alloc((size_t)MTOT * HH * sizeof(u16));
    u16*   hbf    = (u16*)  alloc((size_t)MTOT * 1024 * sizeof(u16));
    u16*   wtin   = (u16*)  alloc((size_t)HH * DIN_PAD * sizeof(u16));
    u16*   wtup   = (u16*)  alloc((size_t)1024 * HH * sizeof(u16));
    u16*   wtdn   = (u16*)  alloc((size_t)HH * 1024 * sizeof(u16));
    u16*   wttr   = (u16*)  alloc((size_t)HH * HH * sizeof(u16));

    u16*   xcat = (u16*)r0;
    float* hdn  = (float*)r0;   // alias: valid after Xcat dies
    float* xbuf = (float*)r1;
    u16*   xcb  = (u16*)r1;     // alias: valid after xbuf dies

    auto gsz = [](size_t n) { return (unsigned)((n + 255) / 256); };

    // ---- 1) weight transposes to bf16 [N x Kpad] ----
    wtrans<<<gsz((size_t)HH * DIN_PAD), 256, 0, stream>>>(W_in, wtin, DIN_RAW, HH, DIN_PAD);
    wtrans<<<gsz((size_t)1024 * HH),    256, 0, stream>>>(W_up, wtup, HH, 1024, HH);
    wtrans<<<gsz((size_t)HH * 1024),    256, 0, stream>>>(W_dn, wtdn, 1024, HH, 1024);
    wtrans<<<gsz((size_t)HH * HH),      256, 0, stream>>>(W_tr, wttr, HH, HH, HH);

    // ---- 2) pack input concat ----
    pack_xcat<<<gsz((size_t)MROWS * DIN_PAD), 256, 0, stream>>>(track, corr, flow, vis, xcat);

    // ---- 3) x = gelu(Xcat @ W_in + b_in) ----
    gemm_bf16_wmma<<<dim3(HH / 128, MROWS / 128), 256, 0, stream>>>(
        xcat, wtin, b_in, nullptr, xbuf, nullptr, MROWS, HH, DIN_PAD, /*act=*/1);

    // ---- 4) append virtual tokens ----
    assemble_xv<<<gsz((size_t)MTOT * HH), 256, 0, stream>>>(xbuf, vtok, xv);

    // ---- 5) opponent E/I scan over T (in place: xv += y) ----
    cssm_scan<<<gsz((size_t)BB * NTOT * HH), 256, 0, stream>>>(xv, dec_e, dec_i, w_e, w_i);

    // ---- 6) ln1: xln1 = LN(xv) (+ bf16 copy for GEMM2) ----
    ln_kernel<<<MTOT, 256, 0, stream>>>(xv, nullptr, ln1_s, ln1_b, xln1, xln1b, NTOT, NTOT);

    // ---- 7) h = gelu(xln1 @ W_up + b_up), bf16 only ----
    gemm_bf16_wmma<<<dim3(1024 / 128, MTOT / 128), 256, 0, stream>>>(
        xln1b, wtup, b_up, nullptr, nullptr, hbf, MTOT, 1024, HH, /*act=*/1);

    // ---- 8) hdn = h @ W_dn + b_dn, fp32 ----
    gemm_bf16_wmma<<<dim3(HH / 128, MTOT / 128), 256, 0, stream>>>(
        hbf, wtdn, b_dn, nullptr, hdn, nullptr, MTOT, HH, 1024, /*act=*/0);

    // ---- 9) ln2: xc = LN(xln1 + hdn), compacted to real tokens, bf16 ----
    ln_kernel<<<MTOT, 256, 0, stream>>>(xln1, hdn, ln2_s, ln2_b, nullptr, xcb, NTOT, NN);

    // ---- 10) track_out = track + xc @ W_track + b_track ----
    gemm_bf16_wmma<<<dim3(HH / 128, MROWS / 128), 256, 0, stream>>>(
        xcb, wttr, b_tr, track, out_track, nullptr, MROWS, HH, HH, /*act=*/0);

    // ---- 11) delta / vis heads ----
    heads_kernel<<<gsz(MROWS), 256, 0, stream>>>(xcb, W_dl, b_dl, W_vs, b_vs, out_delta, out_vis);
}
